// GraphConvolutionSparse_55645596287569
// MI455X (gfx1250) — matrix operand order repaired and checked by reading, not compiled, not run on hardware
//
#include <hip/hip_runtime.h>
#include <hip/hip_bf16.h>

#define NNODES   100000
#define IN_DIM   512
#define OUT_DIM  256

typedef __attribute__((ext_vector_type(16))) __bf16          v16bf;
typedef __attribute__((ext_vector_type(16))) unsigned short  v16us;
typedef __attribute__((ext_vector_type(8)))  float           v8f;

union BFV {
    v16us u;
    v16bf b;
    unsigned short s[16];
};

__device__ __forceinline__ unsigned short f2bf(float f) {
    // round-to-nearest-even f32 -> bf16 (inputs are finite random normals)
    unsigned int u = __float_as_uint(f);
    unsigned int r = u + 0x7FFFu + ((u >> 16) & 1u);
    return (unsigned short)(r >> 16);
}
__device__ __forceinline__ float bf2f(unsigned short s) {
    return __uint_as_float(((unsigned int)s) << 16);
}

__device__ __forceinline__ void atomAddF(float* p, float v) {
    // hardware global_atomic_add_f32 (no CAS loop)
    unsafeAtomicAdd(p, v);
}

// ---------------------------------------------------------------------------
// Utility kernels
// ---------------------------------------------------------------------------
__global__ void zero4_kernel(float4* __restrict__ p, long long n4) {
    long long i = (long long)blockIdx.x * blockDim.x + threadIdx.x;
    if (i < n4) p[i] = make_float4(0.f, 0.f, 0.f, 0.f);
}

__global__ void relu4_kernel(float4* __restrict__ p, long long n4) {
    long long i = (long long)blockIdx.x * blockDim.x + threadIdx.x;
    if (i < n4) {
        float4 v = p[i];
        v.x = fmaxf(v.x, 0.f); v.y = fmaxf(v.y, 0.f);
        v.z = fmaxf(v.z, 0.f); v.w = fmaxf(v.w, 0.f);
        p[i] = v;
    }
}

// ---------------------------------------------------------------------------
// Stage 0: densify sparse features:  Xd[r, c] += val   (5M f32 atomics)
// ---------------------------------------------------------------------------
__global__ void scatter_feat_kernel(const float* __restrict__ vals,
                                    const int*   __restrict__ rows,
                                    const int*   __restrict__ cols,
                                    float* __restrict__ Xd, int nnz) {
    int e = blockIdx.x * blockDim.x + threadIdx.x;
    if (e < nnz)
        atomAddF(&Xd[(size_t)rows[e] * IN_DIM + cols[e]], vals[e]);
}

// ---------------------------------------------------------------------------
// Stage 1: XW = Xd @ W via v_wmma_f32_16x16x32_bf16 with bf16x3 split
// (a_hi+a_lo)(b_hi+b_lo) ~= a_hi*b_hi + a_hi*b_lo + a_lo*b_hi, f32 accumulate
// Block: 256 threads = 8 wave32; tile 128 (M) x 32 (N); K = 512 in 16 chunks.
// ---------------------------------------------------------------------------
__global__ __launch_bounds__(256) void gemm_xw_bf16x3_kernel(
        const float* __restrict__ Xd,
        const float* __restrict__ W,
        float* __restrict__ XW) {
    // B fragments prepacked in LDS: [kchunk 16][ntile 2][lane 32][elem 16]
    __shared__ unsigned short Bhi[16 * 2 * 32 * 16];   // 32 KB
    __shared__ unsigned short Blo[16 * 2 * 32 * 16];   // 32 KB

    const int nBase = blockIdx.x * 32;
    const int mBase = blockIdx.y * 128;

    // ---- prepack W[k, nBase:nBase+32] into bf16 hi/lo B-fragment layout ----
    // B k order used here (same as A): lanes 0-15 = col N=lane, K=0..15 contiguous;
    //                            lanes 16-31 = col N=lane-16, K=16..31.
    for (int slot = threadIdx.x; slot < 16 * 2 * 32; slot += 256) {
        const int lane = slot & 31;
        const int tile = (slot >> 5) & 1;
        const int c    = slot >> 6;
        const int n    = nBase + tile * 16 + (lane & 15);
        const int kb   = c * 32 + ((lane < 16) ? 0 : 16);
        const int base = slot * 16;
#pragma unroll
        for (int j = 0; j < 16; ++j) {
            float w = W[(size_t)(kb + j) * OUT_DIM + n];
            unsigned short h = f2bf(w);
            Bhi[base + j] = h;
            Blo[base + j] = f2bf(w - bf2f(h));
        }
    }
    __syncthreads();

    const int lane = threadIdx.x & 31;
    const int wave = threadIdx.x >> 5;

    // A k order used here (same as B): lane row M = lane%16;
// lanes 0-15 hold K = kc+{0..15};
// lanes 16-31 hold K = kc+16+{0..15}.
    int rowA = mBase + wave * 16 + (lane & 15);
    if (rowA >= NNODES) rowA = NNODES - 1;            // clamp (stores guarded)
    const float* __restrict__ rowp = Xd + (size_t)rowA * IN_DIM;

    v8f acc0 = {};
    v8f acc1 = {};

    for (int c = 0; c < 16; ++c) {
        const int kb = c * 32 + ((lane < 16) ? 0 : 16);
        BFV ahi, alo;
#pragma unroll
        for (int j = 0; j < 8; ++j) {
            float a0 = rowp[kb + j];
            float a1 = rowp[kb + 8 + j];
            unsigned short h0 = f2bf(a0), h1 = f2bf(a1);
            ahi.s[j]     = h0;
            ahi.s[8 + j] = h1;
            alo.s[j]     = f2bf(a0 - bf2f(h0));
            alo.s[8 + j] = f2bf(a1 - bf2f(h1));
        }

        const int bb0 = ((c * 2 + 0) * 32 + lane) * 16;
        const int bb1 = ((c * 2 + 1) * 32 + lane) * 16;
        BFV bh0, bl0, bh1, bl1;
        bh0.u = *(const v16us*)&Bhi[bb0];
        bl0.u = *(const v16us*)&Blo[bb0];
        bh1.u = *(const v16us*)&Bhi[bb1];
        bl1.u = *(const v16us*)&Blo[bb1];

        acc0 = __builtin_amdgcn_wmma_f32_16x16x32_bf16(false, ahi.b, false, bh0.b, (short)0, acc0, false, false);
        acc0 = __builtin_amdgcn_wmma_f32_16x16x32_bf16(false, ahi.b, false, bl0.b, (short)0, acc0, false, false);
        acc0 = __builtin_amdgcn_wmma_f32_16x16x32_bf16(false, alo.b, false, bh0.b, (short)0, acc0, false, false);

        acc1 = __builtin_amdgcn_wmma_f32_16x16x32_bf16(false, ahi.b, false, bh1.b, (short)0, acc1, false, false);
        acc1 = __builtin_amdgcn_wmma_f32_16x16x32_bf16(false, ahi.b, false, bl1.b, (short)0, acc1, false, false);
        acc1 = __builtin_amdgcn_wmma_f32_16x16x32_bf16(false, alo.b, false, bh1.b, (short)0, acc1, false, false);
    }

    // C/D layout: VGPR r, lanes 0-15 -> M=r, lanes 16-31 -> M=8+r; N=lane%16
    const int rowOut = mBase + wave * 16 + ((lane < 16) ? 0 : 8);
    const int colOut = nBase + (lane & 15);
#pragma unroll
    for (int r = 0; r < 8; ++r) {
        const int m = rowOut + r;
        if (m < NNODES) {
            XW[(size_t)m * OUT_DIM + colOut]      = acc0[r];
            XW[(size_t)m * OUT_DIM + colOut + 16] = acc1[r];
        }
    }
}

// ---------------------------------------------------------------------------
// Stage 2: out += adj_val * XW[adj_col, :]   (gather from L2-resident XW,
// scatter with hardware f32 atomics).  64 lanes (64 float4) per edge.
// ---------------------------------------------------------------------------
__global__ __launch_bounds__(256) void spmm_adj_kernel(
        const float* __restrict__ XW,
        const float* __restrict__ vals,
        const int*   __restrict__ rows,
        const int*   __restrict__ cols,
        float* __restrict__ out, int nnz) {
    const int e = blockIdx.x * 4 + (threadIdx.x >> 6);
    if (e >= nnz) return;
    const int q = (threadIdx.x & 63) * 4;
    const int r = rows[e];
    const int c = cols[e];
    const float v = vals[e];
    const float4 x = *(const float4*)&XW[(size_t)c * OUT_DIM + q];
    float* o = out + (size_t)r * OUT_DIM + q;
    atomAddF(o + 0, v * x.x);
    atomAddF(o + 1, v * x.y);
    atomAddF(o + 2, v * x.z);
    atomAddF(o + 3, v * x.w);
}

// ---------------------------------------------------------------------------
extern "C" void kernel_launch(void* const* d_in, const int* in_sizes, int n_in,
                              void* d_out, int out_size, void* d_ws, size_t ws_size,
                              hipStream_t stream) {
    const float* feat_vals = (const float*)d_in[0];
    const float* adj_vals  = (const float*)d_in[1];
    const float* W         = (const float*)d_in[2];
    const int*   feat_rows = (const int*)d_in[3];
    const int*   feat_cols = (const int*)d_in[4];
    const int*   adj_rows  = (const int*)d_in[5];
    const int*   adj_cols  = (const int*)d_in[6];
    float* out = (float*)d_out;

    const int nnzF = in_sizes[0];
    const int nnzA = in_sizes[1];

    float* Xd = (float*)d_ws;                                  // 100000*512 f32
    float* XW = Xd + (size_t)NNODES * IN_DIM;                  // 100000*256 f32

    // zero dense X + XW scratch, and the atomic-accumulated output
    {
        long long n4 = (long long)NNODES * (IN_DIM + OUT_DIM) / 4;   // 19.2M
        int blocks = (int)((n4 + 255) / 256);
        zero4_kernel<<<blocks, 256, 0, stream>>>((float4*)d_ws, n4);

        long long o4 = (long long)NNODES * OUT_DIM / 4;              // 6.4M
        blocks = (int)((o4 + 255) / 256);
        zero4_kernel<<<blocks, 256, 0, stream>>>((float4*)d_out, o4);
    }

    // stage 0: densify sparse features
    scatter_feat_kernel<<<(nnzF + 255) / 256, 256, 0, stream>>>(
        feat_vals, feat_rows, feat_cols, Xd, nnzF);

    // stage 1: XW = Xd @ W  (WMMA bf16x3)
    dim3 g1(OUT_DIM / 32, (NNODES + 127) / 128);                 // (8, 782)
    gemm_xw_bf16x3_kernel<<<g1, 256, 0, stream>>>(Xd, W, XW);

    // stage 2: out = A @ XW (gather + atomic scatter)
    spmm_adj_kernel<<<(nnzA + 3) / 4, 256, 0, stream>>>(
        XW, adj_vals, adj_rows, adj_cols, out, nnzA);

    // stage 3: ReLU in place
    {
        long long o4 = (long long)NNODES * OUT_DIM / 4;
        relu4_kernel<<<(int)((o4 + 255) / 256), 256, 0, stream>>>((float4*)d_out, o4);
    }
}
